// ATOCActorNet_45346264711378
// MI455X (gfx1250) — matrix operand order repaired and checked by reading, not compiled, not run on hardware
//
#include <hip/hip_runtime.h>
#include <hip/hip_bf16.h>
#include <math.h>

// ---------------------------------------------------------------------------
// ATOC actor net for MI455X (gfx1250, wave32, WMMA f16 16x16x32 w/ f32 accum)
// ---------------------------------------------------------------------------
// Dims from reference:
#define A_N   1024
#define OBS_N 2048
#define TD_N  256
#define H_N   128
#define M_N   16
#define ACT_N 256
#define AE_N  64

typedef __attribute__((ext_vector_type(16))) _Float16 v16h;
typedef __attribute__((ext_vector_type(8)))  float    v8f;

__device__ __forceinline__ float sigmf(float x) { return 1.0f / (1.0f + expf(-x)); }

// ---------------------------------------------------------------------------
// Panel packing: convert f32 row-major matrices into WMMA-native f16 layouts.
//
// A-matrix 16x32 f16 layout (ISA 7.12.2): lane L (row = L&15, kbase = L>=16?8:0)
//   half h: v=h>>1, o=h&1 ; K = (v<4) ? kbase+2v+o : 16+kbase+2(v-4)+o
// B-matrix 32x16 f16 layout: lane L (col = L&15, kb = L>=16?8:0); half h: K = (h<8) ? kb+h : 16+kb+h-8
// Panels stored as [tile][lane][half] so GEMM loads are contiguous b128 pairs.
// ---------------------------------------------------------------------------
__global__ void pack_a_kernel(const float* __restrict__ src, _Float16* __restrict__ dst,
                              int Mrows, int K, int relu) {
  int ktiles = K >> 5;
  int total = (Mrows >> 4) * ktiles * 512;
  for (int e = blockIdx.x * blockDim.x + threadIdx.x; e < total;
       e += gridDim.x * blockDim.x) {
    int h = e & 15;
    int L = (e >> 4) & 31;
    int tileL = e >> 9;
    int kt = tileL % ktiles;
    int rt = tileL / ktiles;
    int mrow = (rt << 4) + (L & 15);
    int kbase = (L >= 16) ? 8 : 0;
    int v = h >> 1, o = h & 1;
    int kk = (v < 4) ? (kbase + 2 * v + o) : (16 + kbase + 2 * (v - 4) + o);
    float val = src[(size_t)mrow * K + (kt << 5) + kk];
    if (relu) val = fmaxf(val, 0.0f);
    dst[e] = (_Float16)val;
  }
}

__global__ void pack_b_kernel(const float* __restrict__ src, _Float16* __restrict__ dst,
                              int N, int K, int transposedSrc) {
  int ktiles = K >> 5;
  int total = (N >> 4) * ktiles * 512;
  for (int e = blockIdx.x * blockDim.x + threadIdx.x; e < total;
       e += gridDim.x * blockDim.x) {
    int h = e & 15;
    int L = (e >> 4) & 31;
    int tileL = e >> 9;
    int kt = tileL % ktiles;
    int nt = tileL / ktiles;
    int n = (nt << 4) + (L & 15);
    int kk = (kt << 5) + ((L >= 16) ? 8 : 0) + ((h < 8) ? h : (8 + h));
    float val = transposedSrc ? src[(size_t)n * K + kk]
                              : src[(size_t)kk * N + n];
    dst[e] = (_Float16)val;
  }
}

// ---------------------------------------------------------------------------
// Fused GEMM (M=16 block x N=256) + bias + LayerNorm + activation.
// Block: 512 threads = 16 waves; wave w owns column tile [16w, 16w+16).
// act: 0 = LN only, 1 = LN then ReLU, 2 = LN then tanh.
// ---------------------------------------------------------------------------
__global__ __launch_bounds__(512) void gemm_ln_kernel(
    const _Float16* __restrict__ Apan, const _Float16* __restrict__ Bpan,
    const float* __restrict__ bias, const float* __restrict__ gamma,
    const float* __restrict__ beta, float* __restrict__ out, int K, int act) {
  __shared__ float tile[16][256];
  int tid = threadIdx.x;
  int lane = tid & 31;
  int w = tid >> 5;
  int ktiles = K >> 5;

  v8f acc = {};
  const _Float16* Ap = Apan + ((size_t)blockIdx.x * ktiles) * 512 + lane * 16;
  const _Float16* Bp = Bpan + ((size_t)w * ktiles) * 512 + lane * 16;
  for (int kt = 0; kt < ktiles; ++kt) {
    __builtin_prefetch(Bp + 1024, 0, 1);  // global_prefetch_b8 two tiles ahead
    v16h a = *(const v16h*)Ap;
    v16h b = *(const v16h*)Bp;
    acc = __builtin_amdgcn_wmma_f32_16x16x32_f16(false, a, false, b,
                                                 (short)0, acc, false, false);
    Ap += 512;
    Bp += 512;
  }

  // D layout: reg r, lanes 0-15 -> M=r, lanes 16-31 -> M=r+8; N = lane&15
  int mb = (lane >= 16) ? 8 : 0;
  int n = (w << 4) + (lane & 15);
  float bv = bias[n];
#pragma unroll
  for (int r = 0; r < 8; ++r) tile[r + mb][n] = acc[r] + bv;
  __syncthreads();

  // LayerNorm: wave w normalizes row w; each lane handles 8 consecutive cols.
  int base = lane << 3;
  float s = 0.0f;
#pragma unroll
  for (int i = 0; i < 8; ++i) s += tile[w][base + i];
  for (int off = 16; off; off >>= 1) s += __shfl_xor(s, off);
  float mean = s * (1.0f / 256.0f);
  float vs = 0.0f;
#pragma unroll
  for (int i = 0; i < 8; ++i) {
    float d = tile[w][base + i] - mean;
    vs += d * d;
  }
  for (int off = 16; off; off >>= 1) vs += __shfl_xor(vs, off);
  float rstd = rsqrtf(vs * (1.0f / 256.0f) + 1e-5f);

  float* orow = out + ((size_t)blockIdx.x * 16 + w) * 256;
#pragma unroll
  for (int i = 0; i < 8; ++i) {
    int n2 = base + i;
    float y = (tile[w][n2] - mean) * rstd * gamma[n2] + beta[n2];
    if (act == 1) y = fmaxf(y, 0.0f);
    else if (act == 2) y = tanhf(y);
    orow[n2] = y;
  }
}

// ---------------------------------------------------------------------------
// Attention MLP -> initiator flags. One 64-thread block per agent.
// sigmoid(x) > 0.5  <=>  x > 0, so no exp needed for the decision.
// ---------------------------------------------------------------------------
__global__ __launch_bounds__(64) void attention_kernel(
    const float* __restrict__ thoughts, const float* __restrict__ aW1,
    const float* __restrict__ ab1, const float* __restrict__ aW2,
    const float* __restrict__ ab2, const float* __restrict__ aW3,
    const float* __restrict__ ab3, int* __restrict__ isInit) {
  __shared__ float th[256];
  __shared__ float a1[64];
  __shared__ float red[64];
  int tid = threadIdx.x;
  int row = blockIdx.x;
  const float* trow = thoughts + (size_t)row * 256;
  for (int i = tid; i < 256; i += 64) th[i] = trow[i];
  __syncthreads();

  float s = ab1[tid];
  for (int k = 0; k < 256; ++k) s += th[k] * aW1[k * 64 + tid];
  a1[tid] = fmaxf(s, 0.0f);
  __syncthreads();

  float s2 = ab2[tid];
  for (int k = 0; k < 64; ++k) s2 += a1[k] * aW2[k * 64 + tid];
  float a2 = fmaxf(s2, 0.0f);

  red[tid] = a2 * aW3[tid];
  __syncthreads();
  if (tid < 32) red[tid] += red[tid + 32];
  __syncthreads();
  if (tid == 0) {
    float t = 0.0f;
    for (int i = 0; i < 32; ++i) t += red[i];
    isInit[row] = (t + ab3[0]) > 0.0f ? 1 : 0;
  }
}

// ---------------------------------------------------------------------------
// Serial comm scan. One workgroup of 1024 threads (32 waves).
// Per active step:
//   1. wave32 ballot prefix-scan over the C row -> 16 member indices (sorted)
//   2. gather members, pack f16 A panels (fwd + reversed) into LDS
//   3. WMMA: ZX[16,512] = seq @ Wih^T + (bih+bhh) for both directions
//      (hoists ~2/3 of LSTM FLOPs off the serial path)
//   4. serial 16-step gate recurrence: threads 0-511 fwd, 512-1023 bwd,
//      each thread owns one of the 512 gate pre-activations (128-wide matvec)
//   5. scatter concat(h_f[t], h_b[15-t]) back into thoughts[order[t]]
// ---------------------------------------------------------------------------
__global__ __launch_bounds__(1024) void comm_kernel(
    float* __restrict__ thoughts, const unsigned char* __restrict__ C,
    const int* __restrict__ isInit, const _Float16* __restrict__ WihTfB,
    const _Float16* __restrict__ WihTrB, const float* __restrict__ Whh_f,
    const float* __restrict__ Whh_r, const float* __restrict__ bih_f,
    const float* __restrict__ bhh_f, const float* __restrict__ bih_r,
    const float* __restrict__ bhh_r) {
  __shared__ __align__(32) _Float16 aF[8 * 512];
  __shared__ __align__(32) _Float16 aB[8 * 512];
  __shared__ float ZXf[16][512];
  __shared__ float ZXb[16][512];
  __shared__ float zf[512], zb[512];
  __shared__ float hfh[16][128], hbh[16][128];
  __shared__ float hcur[2][128], ccur[2][128];
  __shared__ float bsum[2][512];
  __shared__ int order[16];
  __shared__ int wcnt[32], woff[32];

  int tid = threadIdx.x;
  int lane = tid & 31;
  int w = tid >> 5;

  if (tid < 512) {
    bsum[0][tid] = bih_f[tid] + bhh_f[tid];
    bsum[1][tid] = bih_r[tid] + bhh_r[tid];
  }
  __syncthreads();

  for (int step = 0; step < A_N; ++step) {
    if (!isInit[step]) continue;  // uniform branch

    // --- 1. member order (indices of the 16 set bits, increasing) ---
    const unsigned char* crow = C + (size_t)step * A_N;
    bool m = crow[tid] != 0;
    unsigned bal = (unsigned)__ballot(m);
    if (lane == 0) wcnt[w] = __popc(bal);
    __syncthreads();
    if (tid == 0) {
      int s = 0;
      for (int i = 0; i < 32; ++i) { woff[i] = s; s += wcnt[i]; }
    }
    __syncthreads();
    if (m) {
      int pos = woff[w] + __popc(bal & ((1u << lane) - 1u));
      if (pos < M_N) order[pos] = tid;
    }
    __syncthreads();

    // --- 2. gather + pack A panels (fwd seq and reversed seq) ---
    for (int e = tid; e < 4096; e += 1024) {
      int kt = e >> 9;
      int L = (e >> 4) & 31;
      int h = e & 15;
      int mrow = L & 15;
      int kbase = (L >= 16) ? 8 : 0;
      int v = h >> 1, o = h & 1;
      int kk = (v < 4) ? (kbase + 2 * v + o) : (16 + kbase + 2 * (v - 4) + o);
      int col = (kt << 5) + kk;
      aF[e] = (_Float16)thoughts[(size_t)order[mrow] * TD_N + col];
      aB[e] = (_Float16)thoughts[(size_t)order[15 - mrow] * TD_N + col];
    }
    __syncthreads();

    // --- 3. ZX = seq @ Wih^T + bias (WMMA): wave w -> column tile w ---
    {
      v8f accF = {};
      v8f accB = {};
      const _Float16* bf = WihTfB + ((size_t)w * 8) * 512 + lane * 16;
      const _Float16* bb = WihTrB + ((size_t)w * 8) * 512 + lane * 16;
#pragma unroll
      for (int kt = 0; kt < 8; ++kt) {
        v16h a1 = *(const v16h*)&aF[kt * 512 + lane * 16];
        v16h b1 = *(const v16h*)bf;
        accF = __builtin_amdgcn_wmma_f32_16x16x32_f16(false, a1, false, b1,
                                                      (short)0, accF, false, false);
        v16h a2 = *(const v16h*)&aB[kt * 512 + lane * 16];
        v16h b2 = *(const v16h*)bb;
        accB = __builtin_amdgcn_wmma_f32_16x16x32_f16(false, a2, false, b2,
                                                      (short)0, accB, false, false);
        bf += 512;
        bb += 512;
      }
      int mb = (lane >= 16) ? 8 : 0;
      int n = (w << 4) + (lane & 15);
#pragma unroll
      for (int r = 0; r < 8; ++r) {
        ZXf[r + mb][n] = accF[r] + bsum[0][n];
        ZXb[r + mb][n] = accB[r] + bsum[1][n];
      }
    }
    if (tid < 256) {
      int g = tid >> 7, j = tid & 127;
      hcur[g][j] = 0.0f;
      ccur[g][j] = 0.0f;
    }
    __syncthreads();

    // --- 4. serial recurrence, both directions concurrently ---
    {
      int g = tid >> 9;       // 0 = forward, 1 = backward
      int j = tid & 511;      // gate pre-activation index
      const float* WhhRow = (g == 0) ? (Whh_f + (size_t)j * H_N)
                                     : (Whh_r + (size_t)j * H_N);
      float(*ZX)[512] = (g == 0) ? ZXf : ZXb;
      float* zbuf = (g == 0) ? zf : zb;
      for (int t = 0; t < M_N; ++t) {
        float z = ZX[t][j];
        for (int k = 0; k < H_N; ++k) z += WhhRow[k] * hcur[g][k];
        zbuf[j] = z;
        __syncthreads();
        if (j < H_N) {
          float zi = zbuf[j], zff = zbuf[128 + j];
          float zg = zbuf[256 + j], zo = zbuf[384 + j];
          float c = sigmf(zff) * ccur[g][j] + sigmf(zi) * tanhf(zg);
          float h = sigmf(zo) * tanhf(c);
          ccur[g][j] = c;
          hcur[g][j] = h;
          if (g == 0) hfh[t][j] = h;
          else hbh[t][j] = h;
        }
        __syncthreads();
      }
    }

    // --- 5. scatter concat(h_f[t], h_b[15-t]) back into thoughts ---
    for (int e = tid; e < 4096; e += 1024) {
      int t = e >> 8;
      int col = e & 255;
      float val = (col < 128) ? hfh[t][col] : hbh[15 - t][col - 128];
      thoughts[(size_t)order[t] * TD_N + col] = val;
    }
    __threadfence();
    __syncthreads();
  }
}

// ---------------------------------------------------------------------------
// Host-side orchestration
// ---------------------------------------------------------------------------
extern "C" void kernel_launch(void* const* d_in, const int* in_sizes, int n_in,
                              void* d_out, int out_size, void* d_ws, size_t ws_size,
                              hipStream_t stream) {
  (void)in_sizes; (void)n_in; (void)out_size; (void)ws_size;

  const float* obs   = (const float*)d_in[0];
  const unsigned char* C = (const unsigned char*)d_in[1];
  const float* W1 = (const float*)d_in[2];
  const float* b1 = (const float*)d_in[3];
  const float* g1 = (const float*)d_in[4];
  const float* bt1 = (const float*)d_in[5];
  const float* W2 = (const float*)d_in[6];
  const float* b2 = (const float*)d_in[7];
  const float* g2 = (const float*)d_in[8];
  const float* bt2 = (const float*)d_in[9];
  const float* aW1 = (const float*)d_in[10];
  const float* ab1 = (const float*)d_in[11];
  const float* aW2 = (const float*)d_in[12];
  const float* ab2 = (const float*)d_in[13];
  const float* aW3 = (const float*)d_in[14];
  const float* ab3 = (const float*)d_in[15];
  const float* Wih_f = (const float*)d_in[16];
  const float* Whh_f = (const float*)d_in[17];
  const float* bih_f = (const float*)d_in[18];
  const float* bhh_f = (const float*)d_in[19];
  const float* Wih_r = (const float*)d_in[20];
  const float* Whh_r = (const float*)d_in[21];
  const float* bih_r = (const float*)d_in[22];
  const float* bhh_r = (const float*)d_in[23];
  const float* W3 = (const float*)d_in[24];
  const float* b3 = (const float*)d_in[25];
  const float* g3 = (const float*)d_in[26];
  const float* bt3 = (const float*)d_in[27];
  const float* W4 = (const float*)d_in[28];
  const float* b4 = (const float*)d_in[29];
  const float* g4 = (const float*)d_in[30];
  const float* bt4 = (const float*)d_in[31];

  // Workspace carve-up (all 256B aligned)
  char* ws = (char*)d_ws;
  size_t off = 0;
  auto carve = [&](size_t bytes) -> char* {
    char* p = ws + off;
    off += (bytes + 255) & ~(size_t)255;
    return p;
  };
  _Float16* obsA   = (_Float16*)carve((size_t)A_N * OBS_N * 2);   // 4 MB
  _Float16* W1B    = (_Float16*)carve((size_t)OBS_N * TD_N * 2);  // 1 MB
  float*    T1     = (float*)   carve((size_t)A_N * TD_N * 4);    // 1 MB
  _Float16* T1A    = (_Float16*)carve((size_t)A_N * TD_N * 2);
  _Float16* W2B    = (_Float16*)carve((size_t)TD_N * TD_N * 2);
  float*    thoughts=(float*)   carve((size_t)A_N * TD_N * 4);
  int*      isInit = (int*)     carve((size_t)A_N * 4);
  _Float16* WihTfB = (_Float16*)carve((size_t)TD_N * 4 * H_N * 2);
  _Float16* WihTrB = (_Float16*)carve((size_t)TD_N * 4 * H_N * 2);
  _Float16* thA    = (_Float16*)carve((size_t)A_N * TD_N * 2);
  _Float16* W3B    = (_Float16*)carve((size_t)TD_N * TD_N * 2);
  float*    t3     = (float*)   carve((size_t)A_N * TD_N * 4);
  _Float16* t3A    = (_Float16*)carve((size_t)A_N * TD_N * 2);
  _Float16* W4B    = (_Float16*)carve((size_t)TD_N * ACT_N * 2);
  float*    acts   = (float*)d_out;

  // --- pack weights + obs into WMMA panels ---
  pack_a_kernel<<<2048, 256, 0, stream>>>(obs, obsA, A_N, OBS_N, 0);
  pack_b_kernel<<<512, 256, 0, stream>>>(W1, W1B, TD_N, OBS_N, 0);
  pack_b_kernel<<<64, 256, 0, stream>>>(W2, W2B, TD_N, TD_N, 0);
  pack_b_kernel<<<64, 256, 0, stream>>>(W3, W3B, TD_N, TD_N, 0);
  pack_b_kernel<<<64, 256, 0, stream>>>(W4, W4B, ACT_N, TD_N, 0);
  pack_b_kernel<<<128, 256, 0, stream>>>(Wih_f, WihTfB, 4 * H_N, TD_N, 1);
  pack_b_kernel<<<128, 256, 0, stream>>>(Wih_r, WihTrB, 4 * H_N, TD_N, 1);

  // --- actor_1 ---
  gemm_ln_kernel<<<A_N / 16, 512, 0, stream>>>(obsA, W1B, b1, g1, bt1, T1,
                                               OBS_N, /*LN+ReLU*/ 1);
  pack_a_kernel<<<512, 256, 0, stream>>>(T1, T1A, A_N, TD_N, 0);
  gemm_ln_kernel<<<A_N / 16, 512, 0, stream>>>(T1A, W2B, b2, g2, bt2, thoughts,
                                               TD_N, /*LN*/ 0);

  // --- attention unit -> initiator flags ---
  attention_kernel<<<A_N, 64, 0, stream>>>(thoughts, aW1, ab1, aW2, ab2, aW3,
                                           ab3, isInit);

  // --- serial comm scan (bi-LSTM over member groups) ---
  comm_kernel<<<1, 1024, 0, stream>>>(thoughts, C, isInit, WihTfB, WihTrB,
                                      Whh_f, Whh_r, bih_f, bhh_f, bih_r, bhh_r);

  // --- actor_2 ---
  pack_a_kernel<<<512, 256, 0, stream>>>(thoughts, thA, A_N, TD_N, /*ReLU*/ 1);
  gemm_ln_kernel<<<A_N / 16, 512, 0, stream>>>(thA, W3B, b3, g3, bt3, t3,
                                               TD_N, /*LN*/ 0);
  pack_a_kernel<<<512, 256, 0, stream>>>(t3, t3A, A_N, TD_N, 0);
  gemm_ln_kernel<<<A_N / 16, 512, 0, stream>>>(t3A, W4B, b4, g4, bt4, acts,
                                               TD_N, /*LN+tanh*/ 2);
}
